// GRUCell_52450140619362
// MI455X (gfx1250) — matrix operand order repaired and checked by reading, not compiled, not run on hardware
//
#include <hip/hip_runtime.h>
#include <hip/hip_bf16.h>
#include <math.h>

#define NUM_ATTRS   128
#define NUM_HIDDENS 512
#define ROWS_TOTAL  (1024 * 128)
#define MROWS       64            // rows per workgroup (4 row-blocks of 16)
#define XS_STRIDE   136           // 128 + 8 pad (bf16) -> conflict-free rows
#define HS_STRIDE   520           // 512 + 8 pad (bf16)
#define SMEM_HALFS  (MROWS * XS_STRIDE + 2 * MROWS * HS_STRIDE)

typedef __attribute__((ext_vector_type(16))) __bf16 v16bf;
typedef __attribute__((ext_vector_type(8)))  float  v8f;

union Frag {
    v16bf v;
    unsigned int u[8];
    uint4 q[2];
};

// Truncating f32 -> bf16 (keep high 16 bits).
__device__ __forceinline__ unsigned int f2bf(float f) {
    return __builtin_bit_cast(unsigned int, f) >> 16;
}
__device__ __forceinline__ float bf2f(unsigned short s) {
    return __builtin_bit_cast(float, (unsigned int)s << 16);
}
// Pack two f32 into bf16x2 with a single v_perm_b32.
__device__ __forceinline__ unsigned int pk2(float a, float b) {
    return __builtin_amdgcn_perm(__builtin_bit_cast(unsigned int, b),
                                 __builtin_bit_cast(unsigned int, a),
                                 0x07060302u);
}

__device__ __forceinline__ v8f wmma_bf16(const v16bf a, const v16bf b, const v8f c) {
    return __builtin_amdgcn_wmma_f32_16x16x32_bf16(
        false, a, false, b, (short)0, c, false, false);
}

// 16x32 bf16 A-fragment from bf16 tile in LDS (row-major, padded stride).
// Lane L (<16): row M=L,    K = kt*32 + {0..7, 16..23}
// Lane L (>=16): row M=L-16, K = kt*32 + {8..15, 24..31}
__device__ __forceinline__ Frag loadA_lds(const unsigned short* base, int lrow,
                                          int stride, int kt, int hi) {
    const unsigned short* p = base + lrow * stride + kt * 32 + hi * 8;
    Frag r;
    r.q[0] = *(const uint4*)(p);        // 8 bf16, 16B aligned
    r.q[1] = *(const uint4*)(p + 16);   // +32B
    return r;
}

// B-fragment: pre-swizzled by pack_w; 8 contiguous dwords per lane.
__device__ __forceinline__ Frag loadB(const unsigned int* __restrict__ P, int nkt,
                                      int ntg, int kt, int lane) {
    const unsigned int* p = P + ((size_t)(ntg * nkt + kt) * 32 + lane) * 8;
    Frag r;
    r.q[0] = *(const uint4*)(p);
    r.q[1] = *(const uint4*)(p + 4);
    return r;
}

__device__ __forceinline__ float sigmoidf_(float x) {
    return 1.0f / (1.0f + __expf(-x));
}
// tanh(x) = 1 - 2/(exp(2x)+1); correct at +/-inf via inf propagation.
__device__ __forceinline__ float tanhf_(float x) {
    return 1.0f - 2.0f / (__expf(2.0f * x) + 1.0f);
}

// ---------------- weight pack: f32 row-major [K x 512] -> bf16 B-fragments ---
__global__ void pack_w(const float* __restrict__ W, unsigned int* __restrict__ P,
                       int K) {
    int t    = blockIdx.x * blockDim.x + threadIdx.x;
    int lane = t & 31;
    int frag = t >> 5;
    int nkt  = K >> 5;
    if (frag >= 32 * nkt) return;
    int nt = frag / nkt;
    int kt = frag % nkt;
    int n  = nt * 16 + (lane & 15);
    int kb = kt * 32 + ((lane >> 4) << 3);
    unsigned int* dst = P + ((size_t)frag * 32 + lane) * 8;
#pragma unroll
    for (int j = 0; j < 8; ++j) {
        float f0 = W[(size_t)(kb + 2 * j + (j >> 2) * 8)     * NUM_HIDDENS + n];
        float f1 = W[(size_t)(kb + 2 * j + (j >> 2) * 8 + 1) * NUM_HIDDENS + n];
        dst[j] = pk2(f0, f1);
    }
}

// ---------------- fused GRU cell ---------------------------------------------
__global__ void __launch_bounds__(256)
gru_kernel(const float* __restrict__ x, const float* __restrict__ hst,
           const float* __restrict__ b_z, const float* __restrict__ b_r,
           const float* __restrict__ b_h,
           const unsigned int* __restrict__ wzp, const unsigned int* __restrict__ wrp,
           const unsigned int* __restrict__ whp, const unsigned int* __restrict__ uzp,
           const unsigned int* __restrict__ urp, const unsigned int* __restrict__ uhp,
           float* __restrict__ out) {
    extern __shared__ unsigned short smem[];
    unsigned short* xs = smem;                           // [64][136] x tile bf16
    unsigned short* hs = smem + MROWS * XS_STRIDE;       // [64][520] h tile bf16
    unsigned short* rh = hs + MROWS * HS_STRIDE;         // [64][520] r*h bf16

    const int tid     = threadIdx.x;
    const int w       = tid >> 5;       // wave 0..7 -> 64-column slice
    const int lane    = tid & 31;
    const int l16     = lane & 15;
    const int hi      = lane >> 4;
    const int rowbase = blockIdx.x * MROWS;

    // ---- cooperative stage: x,h (f32 global) -> bf16 LDS tiles -------------
#pragma unroll
    for (int i = 0; i < 16; ++i) {                 // h: 64 rows x 64 chunks of 8
        int chunk = tid + i * 256;
        int r = chunk >> 6;
        int c = (chunk & 63) * 8;
        const float* p = hst + (size_t)(rowbase + r) * NUM_HIDDENS + c;
        float4 f0 = *(const float4*)p;
        float4 f1 = *(const float4*)(p + 4);
        *(uint4*)(hs + r * HS_STRIDE + c) =
            make_uint4(pk2(f0.x, f0.y), pk2(f0.z, f0.w),
                       pk2(f1.x, f1.y), pk2(f1.z, f1.w));
    }
#pragma unroll
    for (int i = 0; i < 4; ++i) {                  // x: 64 rows x 16 chunks of 8
        int chunk = tid + i * 256;
        int r = chunk >> 4;
        int c = (chunk & 15) * 8;
        const float* p = x + (size_t)(rowbase + r) * NUM_ATTRS + c;
        float4 f0 = *(const float4*)p;
        float4 f1 = *(const float4*)(p + 4);
        *(uint4*)(xs + r * XS_STRIDE + c) =
            make_uint4(pk2(f0.x, f0.y), pk2(f0.z, f0.w),
                       pk2(f1.x, f1.y), pk2(f1.z, f1.w));
    }
    __syncthreads();

    v8f az[4][4], ar[4][4];
    v8f vzero = {};
#pragma unroll
    for (int rb = 0; rb < 4; ++rb)
#pragma unroll
        for (int nt = 0; nt < 4; ++nt) { az[rb][nt] = vzero; ar[rb][nt] = vzero; }

    // ---- pass 1: x@Wz + h@Uz  and  x@Wr + h@Ur -----------------------------
    for (int kt = 0; kt < NUM_ATTRS / 32; ++kt) {
        Frag a[4];
#pragma unroll
        for (int rb = 0; rb < 4; ++rb)
            a[rb] = loadA_lds(xs, rb * 16 + l16, XS_STRIDE, kt, hi);
#pragma unroll
        for (int nt = 0; nt < 4; ++nt) {
            Frag bz = loadB(wzp, NUM_ATTRS / 32, w * 4 + nt, kt, lane);
            Frag br = loadB(wrp, NUM_ATTRS / 32, w * 4 + nt, kt, lane);
#pragma unroll
            for (int rb = 0; rb < 4; ++rb) {
                az[rb][nt] = wmma_bf16(a[rb].v, bz.v, az[rb][nt]);
                ar[rb][nt] = wmma_bf16(a[rb].v, br.v, ar[rb][nt]);
            }
        }
    }
    for (int kt = 0; kt < NUM_HIDDENS / 32; ++kt) {
        Frag a[4];
#pragma unroll
        for (int rb = 0; rb < 4; ++rb)
            a[rb] = loadA_lds(hs, rb * 16 + l16, HS_STRIDE, kt, hi);
#pragma unroll
        for (int nt = 0; nt < 4; ++nt) {
            Frag bz = loadB(uzp, NUM_HIDDENS / 32, w * 4 + nt, kt, lane);
            Frag br = loadB(urp, NUM_HIDDENS / 32, w * 4 + nt, kt, lane);
#pragma unroll
            for (int rb = 0; rb < 4; ++rb) {
                az[rb][nt] = wmma_bf16(a[rb].v, bz.v, az[rb][nt]);
                ar[rb][nt] = wmma_bf16(a[rb].v, br.v, ar[rb][nt]);
            }
        }
    }

    // z = sigmoid(.+bz) kept in az[][]; r*h staged to LDS as bf16
#pragma unroll
    for (int rb = 0; rb < 4; ++rb) {
#pragma unroll
        for (int nt = 0; nt < 4; ++nt) {
            int   col = w * 64 + nt * 16 + l16;
            float bz  = b_z[col];
            float br  = b_r[col];
#pragma unroll
            for (int j = 0; j < 8; ++j) {
                int   lrow = rb * 16 + j + 8 * hi;
                float zv   = sigmoidf_(az[rb][nt][j] + bz);
                float rv   = sigmoidf_(ar[rb][nt][j] + br);
                float hv   = bf2f(hs[lrow * HS_STRIDE + col]);
                az[rb][nt][j] = zv;                       // keep z for combine
                rh[lrow * HS_STRIDE + col] = (unsigned short)f2bf(rv * hv);
            }
        }
    }
    __syncthreads();

    // ---- pass 2: x@Wh + (r*h)@Uh -------------------------------------------
    v8f ah[4][4];
#pragma unroll
    for (int rb = 0; rb < 4; ++rb)
#pragma unroll
        for (int nt = 0; nt < 4; ++nt) ah[rb][nt] = vzero;

    for (int kt = 0; kt < NUM_ATTRS / 32; ++kt) {
        Frag a[4];
#pragma unroll
        for (int rb = 0; rb < 4; ++rb)
            a[rb] = loadA_lds(xs, rb * 16 + l16, XS_STRIDE, kt, hi);
#pragma unroll
        for (int nt = 0; nt < 4; ++nt) {
            Frag bh = loadB(whp, NUM_ATTRS / 32, w * 4 + nt, kt, lane);
#pragma unroll
            for (int rb = 0; rb < 4; ++rb)
                ah[rb][nt] = wmma_bf16(a[rb].v, bh.v, ah[rb][nt]);
        }
    }
    for (int kt = 0; kt < NUM_HIDDENS / 32; ++kt) {
        Frag a[4];
#pragma unroll
        for (int rb = 0; rb < 4; ++rb)
            a[rb] = loadA_lds(rh, rb * 16 + l16, HS_STRIDE, kt, hi);
#pragma unroll
        for (int nt = 0; nt < 4; ++nt) {
            Frag bh = loadB(uhp, NUM_HIDDENS / 32, w * 4 + nt, kt, lane);
#pragma unroll
            for (int rb = 0; rb < 4; ++rb)
                ah[rb][nt] = wmma_bf16(a[rb].v, bh.v, ah[rb][nt]);
        }
    }

    // h_new = h + z*(h_tilde - h)
#pragma unroll
    for (int rb = 0; rb < 4; ++rb) {
#pragma unroll
        for (int nt = 0; nt < 4; ++nt) {
            int   col = w * 64 + nt * 16 + l16;
            float bh  = b_h[col];
#pragma unroll
            for (int j = 0; j < 8; ++j) {
                int   lrow = rb * 16 + j + 8 * hi;
                int   row  = rowbase + lrow;
                float ht   = tanhf_(ah[rb][nt][j] + bh);
                float zv   = az[rb][nt][j];
                float hv   = bf2f(hs[lrow * HS_STRIDE + col]);
                out[(size_t)row * NUM_HIDDENS + col] = hv + zv * (ht - hv);
            }
        }
    }
}

extern "C" void kernel_launch(void* const* d_in, const int* in_sizes, int n_in,
                              void* d_out, int out_size, void* d_ws, size_t ws_size,
                              hipStream_t stream) {
    const float* x   = (const float*)d_in[0];
    const float* hst = (const float*)d_in[1];
    const float* w_z = (const float*)d_in[2];
    const float* u_z = (const float*)d_in[3];
    const float* b_z = (const float*)d_in[4];
    const float* w_r = (const float*)d_in[5];
    const float* u_r = (const float*)d_in[6];
    const float* b_r = (const float*)d_in[7];
    const float* w_h = (const float*)d_in[8];
    const float* u_h = (const float*)d_in[9];
    const float* b_h = (const float*)d_in[10];

    unsigned int* ws  = (unsigned int*)d_ws;
    unsigned int* wzp = ws;
    unsigned int* wrp = ws + 32768;
    unsigned int* whp = ws + 65536;
    unsigned int* uzp = ws + 98304;
    unsigned int* urp = ws + 98304 + 131072;
    unsigned int* uhp = ws + 98304 + 262144;
    (void)ws_size; (void)in_sizes; (void)n_in; (void)out_size;

    pack_w<<<16, 256, 0, stream>>>(w_z, wzp, NUM_ATTRS);
    pack_w<<<16, 256, 0, stream>>>(w_r, wrp, NUM_ATTRS);
    pack_w<<<16, 256, 0, stream>>>(w_h, whp, NUM_ATTRS);
    pack_w<<<64, 256, 0, stream>>>(u_z, uzp, NUM_HIDDENS);
    pack_w<<<64, 256, 0, stream>>>(u_r, urp, NUM_HIDDENS);
    pack_w<<<64, 256, 0, stream>>>(u_h, uhp, NUM_HIDDENS);

    gru_kernel<<<ROWS_TOTAL / MROWS, 256, SMEM_HALFS * sizeof(unsigned short),
                 stream>>>(x, hst, b_z, b_r, b_h, wzp, wrp, whp, uzp, urp, uhp,
                           (float*)d_out);
}
